// BYOGPT_20435454394569
// MI455X (gfx1250) — matrix operand order repaired and checked by reading, not compiled, not run on hardware
//
#include <hip/hip_runtime.h>
#include <hip/hip_bf16.h>
#include <math.h>

typedef __attribute__((ext_vector_type(16))) _Float16 v16h;
typedef __attribute__((ext_vector_type(8)))  float    v8f;

#define DMODEL  768
#define SLEN    1024
#define NHEADS  12
#define DHEAD   64
#define NLAYERS 4
#define BATCH   2
#define MROWS   (BATCH * SLEN)
#define VOCABSZ 50257

// ---------------------------------------------------------------------------
// WMMA fragment loaders (wave32, contiguous K order, shared by A and B)
// A-matrix 16x32 f16: lanes 0-15 = rows 0-15 holding K 0..15,
//                     lanes 16-31 = rows 0-15 holding K 16..31.
// B-matrix 32x16 f16: lane(&15) = N column; halves = 16 contiguous K values,
//                     lanes 0-15 cover K 0..15, lanes 16-31 cover K 16..31.
// C/D 16x16 f32: VGPR r: lanes 0-15 -> row r, lanes 16-31 -> row r+8, N=lane&15.
// ---------------------------------------------------------------------------

static __device__ __forceinline__ v16h ld_afrag_f32(const float* __restrict__ base, int ld)
{
  const int lane = threadIdx.x & 31;
  const int r  = lane & 15;
  const int ko = (lane < 16) ? 0 : 16;
  const float* p = base + (size_t)r * ld + ko;
  v16h f;
#pragma unroll
  for (int i = 0; i < 8; ++i) f[i]     = (_Float16)p[i];
#pragma unroll
  for (int i = 0; i < 8; ++i) f[8 + i] = (_Float16)p[i + 8];
  return f;
}

// Unguarded B fragment: caller guarantees all 16 rows valid.
static __device__ __forceinline__ v16h ld_bfrag_f32(const float* __restrict__ base, int ld)
{
  const int lane = threadIdx.x & 31;
  const int n  = lane & 15;
  const int ko = (lane < 16) ? 0 : 16;
  const float* p = base + (size_t)n * ld + ko;
  v16h f;
#pragma unroll
  for (int i = 0; i < 16; ++i) f[i] = (_Float16)p[i];
  return f;
}

// Row-bounded B fragment (edge tiles only).
static __device__ __forceinline__ v16h ld_bfrag_f32_chk(const float* __restrict__ base, int ld, int rows_valid)
{
  const int lane = threadIdx.x & 31;
  const int n  = lane & 15;
  const int ko = (lane < 16) ? 0 : 16;
  v16h f;
  if (n < rows_valid) {
    const float* p = base + (size_t)n * ld + ko;
#pragma unroll
    for (int i = 0; i < 16; ++i) f[i] = (_Float16)p[i];
  } else {
#pragma unroll
    for (int i = 0; i < 16; ++i) f[i] = (_Float16)0.f;
  }
  return f;
}

static __device__ __forceinline__ v16h ld_bfrag_f16(const _Float16* __restrict__ base, int ld)
{
  const int lane = threadIdx.x & 31;
  const int n  = lane & 15;
  const int ko = (lane < 16) ? 0 : 16;
  const _Float16* p = base + (size_t)n * ld + ko;
  v16h f;
#pragma unroll
  for (int i = 0; i < 16; ++i) f[i] = p[i];
  return f;
}

// ---------------------------------------------------------------------------
// Generic GEMM: C[M,N] = A[M,K] @ W[N,K]^T + bias[N]   (torch Linear layout)
// One wave per 16 x (NT*16) output tile; NT tiles reuse one A fragment.
// CHECK=false: N guaranteed divisible by NT*16 (no guards anywhere).
// CHECK=true : wave-uniform full/partial split hoisted out of the k-loop.
// ---------------------------------------------------------------------------
template <int NT, bool CHECK>
__global__ __launch_bounds__(128) void gemm_wmma_kernel(
    const float* __restrict__ A, const float* __restrict__ W,
    const float* __restrict__ bias, float* __restrict__ C,
    int M, int N, int K)
{
  const int wave = threadIdx.x >> 5;
  const int lane = threadIdx.x & 31;
  const int m0 = blockIdx.y * 64 + wave * 16;
  const int n0 = blockIdx.x * (NT * 16);
  if (m0 >= M) return;

  v8f acc[NT];
#pragma unroll
  for (int t = 0; t < NT; ++t) acc[t] = {};

  const float* arow = A + (size_t)m0 * K;
  const bool full = !CHECK || (n0 + NT * 16 <= N);

  if (full) {
    for (int k = 0; k < K; k += 32) {
      __builtin_prefetch(arow + k + 32, 0, 3);
      const v16h a = ld_afrag_f32(arow + k, K);
#pragma unroll
      for (int t = 0; t < NT; ++t) {
        const v16h b = ld_bfrag_f32(W + (size_t)(n0 + t * 16) * K + k, K);
        acc[t] = __builtin_amdgcn_wmma_f32_16x16x32_f16(false, a, false, b, (short)0, acc[t], false, false);
      }
    }
  } else {
    for (int k = 0; k < K; k += 32) {
      const v16h a = ld_afrag_f32(arow + k, K);
#pragma unroll
      for (int t = 0; t < NT; ++t) {
        const v16h b = ld_bfrag_f32_chk(W + (size_t)(n0 + t * 16) * K + k, K, N - n0 - t * 16);
        acc[t] = __builtin_amdgcn_wmma_f32_16x16x32_f16(false, a, false, b, (short)0, acc[t], false, false);
      }
    }
  }

  const int nc   = lane & 15;
  const int roff = (lane < 16) ? 0 : 8;
  if (full) {
#pragma unroll
    for (int t = 0; t < NT; ++t) {
      const int col = n0 + t * 16 + nc;
      const float bv = bias[col];
#pragma unroll
      for (int r = 0; r < 8; ++r)
        C[(size_t)(m0 + r + roff) * N + col] = acc[t][r] + bv;
    }
  } else {
#pragma unroll
    for (int t = 0; t < NT; ++t) {
      const int col = n0 + t * 16 + nc;
      if (col < N) {
        const float bv = bias[col];
#pragma unroll
        for (int r = 0; r < 8; ++r)
          C[(size_t)(m0 + r + roff) * N + col] = acc[t][r] + bv;
      }
    }
  }
}

// ---------------------------------------------------------------------------
// Flash attention: one wave per (16-row q-tile, head, batch).
// Streams key blocks of 32 with online softmax; WMMA for QK^T and P*V.
// Q,K are f32 [B*S, 768] (head = column slice); Vt is f16 [B,H,Dh,S].
// ---------------------------------------------------------------------------
__global__ __launch_bounds__(32) void flash_attn_kernel(
    const float* __restrict__ Q, const float* __restrict__ Kmat,
    const _Float16* __restrict__ Vt, float* __restrict__ O)
{
  __shared__ float sS[16][32];
  __shared__ float sM[16], sL[16], sA[16];

  const int lane = threadIdx.x & 31;
  const int q0 = blockIdx.x * 16;
  const int h  = blockIdx.y;
  const int b  = blockIdx.z;
  const float scale = 0.125f;   // 1/sqrt(64)

  if (lane < 16) { sM[lane] = -__builtin_inff(); sL[lane] = 0.f; }
  __syncthreads();

  const float* qbase = Q + ((size_t)(b * SLEN + q0)) * DMODEL + h * DHEAD;
  const v16h aQ0 = ld_afrag_f32(qbase,      DMODEL);   // dh 0..31
  const v16h aQ1 = ld_afrag_f32(qbase + 32, DMODEL);   // dh 32..63

  v8f o0 = {}, o1 = {}, o2 = {}, o3 = {};

  const int nc   = lane & 15;
  const int roff = (lane < 16) ? 0 : 8;
  const int nblocks = (q0 + 16 + 31) / 32;             // causal cut

  for (int blk = 0; blk < nblocks; ++blk) {
    const int kb = blk * 32;
    const float* kbase = Kmat + ((size_t)(b * SLEN + kb)) * DMODEL + h * DHEAD;

    // scores: Q[16x64] x K_blk^T[64x32] as two 16x16 tiles, 2 WMMA each
    v8f c0 = {}, c1 = {};
    v16h bK;
    bK = ld_bfrag_f32(kbase, DMODEL);
    c0 = __builtin_amdgcn_wmma_f32_16x16x32_f16(false, aQ0, false, bK, (short)0, c0, false, false);
    bK = ld_bfrag_f32(kbase + 32, DMODEL);
    c0 = __builtin_amdgcn_wmma_f32_16x16x32_f16(false, aQ1, false, bK, (short)0, c0, false, false);
    bK = ld_bfrag_f32(kbase + (size_t)16 * DMODEL, DMODEL);
    c1 = __builtin_amdgcn_wmma_f32_16x16x32_f16(false, aQ0, false, bK, (short)0, c1, false, false);
    bK = ld_bfrag_f32(kbase + (size_t)16 * DMODEL + 32, DMODEL);
    c1 = __builtin_amdgcn_wmma_f32_16x16x32_f16(false, aQ1, false, bK, (short)0, c1, false, false);

    // scale + causal mask -> LDS
#pragma unroll
    for (int r = 0; r < 8; ++r) {
      const int row = q0 + r + roff;
      sS[r + roff][nc]      = (kb + nc      <= row) ? c0[r] * scale : -__builtin_inff();
      sS[r + roff][nc + 16] = (kb + 16 + nc <= row) ? c1[r] * scale : -__builtin_inff();
    }
    __syncthreads();

    // online softmax: one lane per row
    if (lane < 16) {
      const float mprev = sM[lane];
      float mx = mprev;
      for (int j = 0; j < 32; ++j) mx = fmaxf(mx, sS[lane][j]);
      const float al = __expf(mprev - mx);
      float sum = 0.f;
      for (int j = 0; j < 32; ++j) {
        const float p = __expf(sS[lane][j] - mx);
        sS[lane][j] = p;
        sum += p;
      }
      sA[lane] = al;
      sM[lane] = mx;
      sL[lane] = sL[lane] * al + sum;
    }
    __syncthreads();

    // rescale accumulators by per-row alpha
#pragma unroll
    for (int r = 0; r < 8; ++r) {
      const float al = sA[r + roff];
      o0[r] *= al; o1[r] *= al; o2[r] *= al; o3[r] *= al;
    }

    // P fragment (A layout) from LDS, converted to f16
    v16h aP;
    {
      const int pr = lane & 15;
      const int pk = (lane < 16) ? 0 : 16;
#pragma unroll
      for (int i = 0; i < 8; ++i) aP[i]     = (_Float16)sS[pr][pk + i];
#pragma unroll
      for (int i = 0; i < 8; ++i) aP[8 + i] = (_Float16)sS[pr][pk + 8 + i];
    }

    // P[16x32] x V_blk[32x64]: 4 WMMA over dh tiles, Vt gives contiguous keys
    const _Float16* vb = Vt + ((size_t)((b * NHEADS + h) * DHEAD)) * SLEN + kb;
    v16h bV;
    bV = ld_bfrag_f16(vb + (size_t) 0 * SLEN, SLEN);
    o0 = __builtin_amdgcn_wmma_f32_16x16x32_f16(false, aP, false, bV, (short)0, o0, false, false);
    bV = ld_bfrag_f16(vb + (size_t)16 * SLEN, SLEN);
    o1 = __builtin_amdgcn_wmma_f32_16x16x32_f16(false, aP, false, bV, (short)0, o1, false, false);
    bV = ld_bfrag_f16(vb + (size_t)32 * SLEN, SLEN);
    o2 = __builtin_amdgcn_wmma_f32_16x16x32_f16(false, aP, false, bV, (short)0, o2, false, false);
    bV = ld_bfrag_f16(vb + (size_t)48 * SLEN, SLEN);
    o3 = __builtin_amdgcn_wmma_f32_16x16x32_f16(false, aP, false, bV, (short)0, o3, false, false);
    __syncthreads();   // sS reused next block
  }

  float* obase = O + ((size_t)(b * SLEN + q0)) * DMODEL + h * DHEAD;
#pragma unroll
  for (int r = 0; r < 8; ++r) {
    const float inv = 1.f / sL[r + roff];
    obase[(size_t)(r + roff) * DMODEL + nc]      = o0[r] * inv;
    obase[(size_t)(r + roff) * DMODEL + 16 + nc] = o1[r] * inv;
    obase[(size_t)(r + roff) * DMODEL + 32 + nc] = o2[r] * inv;
    obase[(size_t)(r + roff) * DMODEL + 48 + nc] = o3[r] * inv;
  }
}

// ---------------------------------------------------------------------------
// Elementwise / reduction helpers
// ---------------------------------------------------------------------------
__global__ __launch_bounds__(256) void embed_kernel(
    const int* __restrict__ tok, const float* __restrict__ emb,
    const float* __restrict__ pe, float* __restrict__ X)
{
  const int idx = blockIdx.x * 256 + threadIdx.x;
  if (idx < MROWS * DMODEL) {
    const int m = idx / DMODEL, c = idx - m * DMODEL;
    const int s = m & (SLEN - 1);
    X[idx] = emb[(size_t)tok[m] * DMODEL + c] + pe[(size_t)s * DMODEL + c];
  }
}

__global__ __launch_bounds__(256) void vtrans_kernel(
    const float* __restrict__ V, _Float16* __restrict__ Vt)
{
  const int idx = blockIdx.x * 256 + threadIdx.x;
  if (idx < MROWS * DMODEL) {
    const int m = idx / DMODEL, c = idx - m * DMODEL;
    const int b = m / SLEN, s = m - b * SLEN;
    const int h = c / DHEAD, d = c - h * DHEAD;
    Vt[((size_t)((b * NHEADS + h) * DHEAD + d)) * SLEN + s] = (_Float16)V[idx];
  }
}

// X = LayerNorm(X + R) * sc + bi, one 256-thread block per row (D = 768 = 3*256)
__global__ __launch_bounds__(256) void add_ln_kernel(
    float* __restrict__ X, const float* __restrict__ R,
    const float* __restrict__ sc, const float* __restrict__ bi)
{
  __shared__ float sred[256];
  const int row = blockIdx.x;
  const int tid = threadIdx.x;
  float* xr = X + (size_t)row * DMODEL;
  const float* rr = R + (size_t)row * DMODEL;

  const float v0 = xr[tid]       + rr[tid];
  const float v1 = xr[tid + 256] + rr[tid + 256];
  const float v2 = xr[tid + 512] + rr[tid + 512];

  sred[tid] = v0 + v1 + v2;
  __syncthreads();
  for (int off = 128; off > 0; off >>= 1) {
    if (tid < off) sred[tid] += sred[tid + off];
    __syncthreads();
  }
  const float mean = sred[0] * (1.0f / DMODEL);
  __syncthreads();

  const float d0 = v0 - mean, d1 = v1 - mean, d2 = v2 - mean;
  sred[tid] = d0 * d0 + d1 * d1 + d2 * d2;
  __syncthreads();
  for (int off = 128; off > 0; off >>= 1) {
    if (tid < off) sred[tid] += sred[tid + off];
    __syncthreads();
  }
  const float var = sred[0] * (1.0f / DMODEL);
  const float rs = rsqrtf(var + 1e-5f);

  xr[tid]       = d0 * rs * sc[tid]       + bi[tid];
  xr[tid + 256] = d1 * rs * sc[tid + 256] + bi[tid + 256];
  xr[tid + 512] = d2 * rs * sc[tid + 512] + bi[tid + 512];
}

// ---------------------------------------------------------------------------
// Launch: embed -> 4x [QKV proj, attn, LN1, lin, LN2] -> unembed
// Workspace: 5 f32 activation buffers + 1 f16 Vt buffer (~35 MB total)
// ---------------------------------------------------------------------------
extern "C" void kernel_launch(void* const* d_in, const int* in_sizes, int n_in,
                              void* d_out, int out_size, void* d_ws, size_t ws_size,
                              hipStream_t stream)
{
  (void)in_sizes; (void)n_in; (void)out_size; (void)ws_size;

  const int*   tokens  = (const int*)  d_in[0];
  const float* embedw  = (const float*)d_in[1];
  const float* pe      = (const float*)d_in[2];
  const float* wq_w    = (const float*)d_in[3];
  const float* wq_b    = (const float*)d_in[4];
  const float* wk_w    = (const float*)d_in[5];
  const float* wk_b    = (const float*)d_in[6];
  const float* wv_w    = (const float*)d_in[7];
  const float* wv_b    = (const float*)d_in[8];
  const float* lin_w   = (const float*)d_in[9];
  const float* lin_b   = (const float*)d_in[10];
  const float* n1_s    = (const float*)d_in[11];
  const float* n1_b    = (const float*)d_in[12];
  const float* n2_s    = (const float*)d_in[13];
  const float* n2_b    = (const float*)d_in[14];
  const float* unemb_w = (const float*)d_in[15];
  const float* unemb_b = (const float*)d_in[16];
  float* logits = (float*)d_out;

  const size_t MD = (size_t)MROWS * DMODEL;
  char* ws = (char*)d_ws;
  float*    x    = (float*)ws;    ws += MD * sizeof(float);
  float*    q    = (float*)ws;    ws += MD * sizeof(float);
  float*    kk   = (float*)ws;    ws += MD * sizeof(float);
  float*    vv   = (float*)ws;    ws += MD * sizeof(float);
  float*    attn = (float*)ws;    ws += MD * sizeof(float);
  _Float16* vt   = (_Float16*)ws; ws += MD * sizeof(_Float16);

  const dim3 eg((MROWS * DMODEL + 255) / 256);
  embed_kernel<<<eg, 256, 0, stream>>>(tokens, embedw, pe, x);

  const dim3 gproj(DMODEL / 64, MROWS / 64);      // 12 x 32 blocks, 4 waves each
  const dim3 gflash(SLEN / 16, NHEADS, BATCH);    // 64 x 12 x 2 waves

  for (int l = 0; l < NLAYERS; ++l) {
    const size_t wo = (size_t)l * DMODEL * DMODEL;
    const size_t bo = (size_t)l * DMODEL;
    gemm_wmma_kernel<4, false><<<gproj, 128, 0, stream>>>(x, wq_w + wo, wq_b + bo, q,  MROWS, DMODEL, DMODEL);
    gemm_wmma_kernel<4, false><<<gproj, 128, 0, stream>>>(x, wk_w + wo, wk_b + bo, kk, MROWS, DMODEL, DMODEL);
    gemm_wmma_kernel<4, false><<<gproj, 128, 0, stream>>>(x, wv_w + wo, wv_b + bo, vv, MROWS, DMODEL, DMODEL);
    vtrans_kernel<<<eg, 256, 0, stream>>>(vv, vt);
    flash_attn_kernel<<<gflash, 32, 0, stream>>>(q, kk, vt, attn);
    add_ln_kernel<<<MROWS, 256, 0, stream>>>(x, attn, n1_s + bo, n1_b + bo);
    gemm_wmma_kernel<4, false><<<gproj, 128, 0, stream>>>(x, lin_w + wo, lin_b + bo, q, MROWS, DMODEL, DMODEL);
    add_ln_kernel<<<MROWS, 256, 0, stream>>>(x, q, n2_s + bo, n2_b + bo);
  }

  // Unembed: 16x128 tiles per wave for WMMA density on the dominant GEMM.
  const dim3 gun((VOCABSZ + 127) / 128, MROWS / 64);
  gemm_wmma_kernel<8, true><<<gun, 128, 0, stream>>>(x, unemb_w, unemb_b, logits, MROWS, VOCABSZ, DMODEL);
}
